// RoiGridPool_42417097016363
// MI455X (gfx1250) — matrix operand order repaired and checked by reading, not compiled, not run on hardware
//
#include <hip/hip_runtime.h>

#define DEV __device__ __forceinline__

typedef __attribute__((ext_vector_type(16))) _Float16 v16h;
typedef __attribute__((ext_vector_type(8)))  float    v8f;
typedef __attribute__((ext_vector_type(4)))  unsigned int v4u;

union FragU { v16h h; v4u u[2]; };

static constexpr int BB  = 2;      // batch
static constexpr int NN  = 100;    // proposals
static constexpr int MM  = 216;    // grid samples per proposal
static constexpr int KK  = 2048;   // keypoints
static constexpr int CC  = 128;    // keypoint channels
static constexpr int NP  = NN * MM;      // 21600 points per batch
static constexpr int GT  = BB * NP;      // 43200 total gridpoints
static constexpr int KP1 = 160;          // 3+128 padded to mult of 32
static constexpr int KR  = CC * MM;      // 27648 reduction K
static constexpr int ORD = 256;          // reduction width
static constexpr int MR  = BB * NN;      // 200 reduction rows
static constexpr int MRP = 208;          // padded to 13 M-tiles

// ---- WMMA fragment helpers (CDNA5 16x16x32 f16 layouts, wave32) ------------
// A (16x32, 16-bit): lane row = lane&15, hi = lane>>4.
//   VGPR j=0..3 -> K = hi*8 + 2j      (contig halfs [hi*8,   hi*8+8) )
//   VGPR j=4..7 -> K = 16 + hi*8 + 2(j-4) (contig halfs [16+hi*8, +8)    )
DEV v16h load_a_frag(const _Float16* base, int rowBase, int strideH, int kBase, int lane) {
  const int r  = rowBase + (lane & 15);
  const int hi = lane >> 4;
  const _Float16* p = base + (size_t)r * strideH + kBase + hi * 8;
  FragU f;
  f.u[0] = *(const v4u*)(p);
  f.u[1] = *(const v4u*)(p + 16);
  return f.h;
}
// B (32x16, 16-bit): lane col = lane&15, hi = lane>>4.
//   VGPR j=0..3 -> K = hi*8 + 2j, j=4..7 -> K = 16 + hi*8 + 2(j-4)  (same as A)
// Stored column-major (== weight (out,in) row-major).
DEV v16h load_b_frag(const _Float16* base, int colBase, int strideH, int kBase, int lane) {
  const int c  = colBase + (lane & 15);
  const int hi = lane >> 4;
  const _Float16* p = base + (size_t)c * strideH + kBase + hi * 8;
  FragU f;
  f.u[0] = *(const v4u*)(p);
  f.u[1] = *(const v4u*)(p + 16);
  return f.h;
}
DEV v8f wmma_f16(v16h a, v16h b, v8f c) {
  return __builtin_amdgcn_wmma_f32_16x16x32_f16(false, a, false, b, (short)0, c, false, false);
}

// ---- 1) gridpoint generation ----------------------------------------------
__global__ __launch_bounds__(256) void gridpoints_kernel(
    const float* __restrict__ proposals, const float* __restrict__ noise,
    float* __restrict__ gridpoints) {
  int g = blockIdx.x * 256 + threadIdx.x;
  if (g >= GT) return;
  int bb = g / NP, rem = g - bb * NP;
  int p = rem / MM;
  const float* pr = proposals + (size_t)(bb * NN + p) * 7;
  const float* nz = noise + (size_t)g * 3;
  float x = pr[3] * (nz[0] - 0.5f);
  float y = pr[4] * (nz[1] - 0.5f);
  float z = pr[5] * (nz[2] - 0.5f);
  float ct = __cosf(pr[6]), st = __sinf(pr[6]);
  gridpoints[(size_t)g * 3 + 0] = pr[0] + ct * x - st * y;
  gridpoints[(size_t)g * 3 + 1] = pr[1] + st * x + ct * y;
  gridpoints[(size_t)g * 3 + 2] = pr[2] + z;
}

// ---- 2) f32 -> f16 weight conversion --------------------------------------
__global__ __launch_bounds__(256) void cvt_kernel(
    const float* __restrict__ src, _Float16* __restrict__ dst, int n) {
  int stride = gridDim.x * 256;
  for (int i = blockIdx.x * 256 + threadIdx.x; i < n; i += stride)
    dst[i] = (_Float16)src[i];
}

// ---- 3) ball query, both radii in one pass --------------------------------
__global__ __launch_bounds__(256) void ball_query_kernel(
    const float* __restrict__ gridpoints, const float* __restrict__ kxyz,
    int* __restrict__ idx0, int* __restrict__ idx1) {
  __shared__ float sK[512 * 3];
  const int tid = threadIdx.x;
  const int bb = blockIdx.y;
  const int np = blockIdx.x * 256 + tid;
  const bool valid = np < NP;
  const int g = bb * NP + np;
  float px = 0.f, py = 0.f, pz = 0.f;
  if (valid) {
    px = gridpoints[(size_t)g * 3 + 0];
    py = gridpoints[(size_t)g * 3 + 1];
    pz = gridpoints[(size_t)g * 3 + 2];
  }
  int cnt0 = 0, cnt1 = 0, first0 = 0, first1 = 0;
  const float r0sq = 0.8f * 0.8f, r1sq = 1.6f * 1.6f;
  const float* kx = kxyz + (size_t)bb * KK * 3;
  for (int cb = 0; cb < KK; cb += 512) {
    __syncthreads();
    for (int i = tid; i < 512 * 3; i += 256) sK[i] = kx[cb * 3 + i];
    __syncthreads();
    if (valid && !(cnt0 == 16 && cnt1 == 16)) {
      for (int kk = 0; kk < 512; ++kk) {
        float dx = sK[kk * 3 + 0] - px;
        float dy = sK[kk * 3 + 1] - py;
        float dz = sK[kk * 3 + 2] - pz;
        float d2 = dx * dx + dy * dy + dz * dz;
        if (d2 < r1sq) {
          int kidx = cb + kk;
          if (cnt1 < 16) { if (cnt1 == 0) first1 = kidx; idx1[(size_t)g * 16 + cnt1] = kidx; ++cnt1; }
          if (d2 < r0sq && cnt0 < 16) { if (cnt0 == 0) first0 = kidx; idx0[(size_t)g * 16 + cnt0] = kidx; ++cnt0; }
        }
      }
    }
  }
  if (valid) {
    for (int j = cnt0; j < 16; ++j) idx0[(size_t)g * 16 + j] = first0;
    for (int j = cnt1; j < 16; ++j) idx1[(size_t)g * 16 + j] = first1;
  }
}

// ---- 4) gather + shared-MLP (2 layers, WMMA) + neighbor max-pool ----------
// WG handles 4 gridpoints => 64 rows (point-major, 16 samples each).
// M-tile w covers exactly one gridpoint's 16 samples -> pool stays in-fragment.
__global__ __launch_bounds__(256) void group_mlp_kernel(
    const float* __restrict__ gridpoints, const float* __restrict__ kxyz,
    const float* __restrict__ kfeat, const int* __restrict__ idxS,
    const float* __restrict__ W1, const float* __restrict__ W2,
    _Float16* __restrict__ featbuf, int chanOfs) {
  __shared__ __align__(16) _Float16 sA[64 * KP1];   // gathered rows (K padded)
  __shared__ __align__(16) _Float16 sB1[64 * KP1];  // W1 col-major(K) == row-major(o,c)
  __shared__ __align__(16) _Float16 sB2[64 * 64];
  __shared__ __align__(16) _Float16 sH[64 * 64];    // hidden activations
  __shared__ int   sIdx[64];
  __shared__ float sGP[12];

  const int tid = threadIdx.x;
  const int g0  = blockIdx.x * 4;
  const int bb  = g0 / NP;

  if (tid < 64) sIdx[tid] = idxS[(size_t)g0 * 16 + tid];
  if (tid < 12) sGP[tid]  = gridpoints[(size_t)g0 * 3 + tid];
  for (int i = tid; i < 64 * KP1; i += 256) {
    int n = i / KP1, k = i - n * KP1;
    sB1[i] = (k < 131) ? (_Float16)W1[n * 131 + k] : (_Float16)0.f;
  }
  for (int i = tid; i < 64 * 64; i += 256) sB2[i] = (_Float16)W2[i];
  __syncthreads();

  // gather: row = p_local*16 + s ; cols = [gxyz(3) | kfeat(128) | pad(29)]
  const float* kx = kxyz + (size_t)bb * KK * 3;
  const float* kf = kfeat + (size_t)bb * CC * KK;
  for (int i = tid; i < 64 * KP1; i += 256) {
    int r = i / KP1, k = i - r * KP1;
    int idx = sIdx[r];
    float v;
    if (k < 3)        v = kx[idx * 3 + k] - sGP[(r >> 4) * 3 + k];
    else if (k < 131) v = kf[(k - 3) * KK + idx];
    else              v = 0.f;
    sA[i] = (_Float16)v;
  }
  __syncthreads();

  const int wave = tid >> 5, lane = tid & 31;
  const int mt = wave & 3;             // M-tile (== local gridpoint)
  const int nt0 = (wave >> 2) * 2;     // this wave's 2 N-tiles
  const int hi = lane >> 4, col = lane & 15;

  // layer 1: (64 x 160) @ (160 x 64)
  for (int t = 0; t < 2; ++t) {
    int nt = nt0 + t;
    v8f acc = {};
#pragma unroll
    for (int ks = 0; ks < 5; ++ks) {
      v16h a = load_a_frag(sA,  mt * 16, KP1, ks * 32, lane);
      v16h b = load_b_frag(sB1, nt * 16, KP1, ks * 32, lane);
      acc = wmma_f16(a, b, acc);
    }
#pragma unroll
    for (int v = 0; v < 8; ++v) {
      int rl = mt * 16 + v + 8 * hi;
      sH[rl * 64 + nt * 16 + col] = (_Float16)fmaxf(acc[v], 0.f);
    }
  }
  __syncthreads();

  // layer 2 + max over the 16 samples of this tile
  for (int t = 0; t < 2; ++t) {
    int nt = nt0 + t;
    v8f acc = {};
#pragma unroll
    for (int ks = 0; ks < 2; ++ks) {
      v16h a = load_a_frag(sH,  mt * 16, 64, ks * 32, lane);
      v16h b = load_b_frag(sB2, nt * 16, 64, ks * 32, lane);
      acc = wmma_f16(a, b, acc);
    }
    float mx = 0.f;  // relu(x) >= 0 so 0-init == relu-then-max
#pragma unroll
    for (int v = 0; v < 8; ++v) mx = fmaxf(mx, acc[v]);
    mx = fmaxf(mx, __shfl_xor(mx, 16, 32));  // combine M and M+8 halves
    if (lane < 16)
      featbuf[(size_t)(g0 + mt) * CC + chanOfs + nt * 16 + col] = (_Float16)mx;
  }
}

// ---- 5) reduction MLP layer 1: (200 x 27648) @ (27648 x 256) --------------
// One WG per M-tile; 8 waves x 2 N-tiles; A staged 16x256 per K-block.
__global__ __launch_bounds__(256) void reduce_mlp1_kernel(
    const _Float16* __restrict__ featbuf, const _Float16* __restrict__ Wr1h,
    _Float16* __restrict__ out1h) {
  __shared__ __align__(16) _Float16 sA2[16 * 256];
  const int tid = threadIdx.x;
  const int mt = blockIdx.x;  // 0..12
  const int wave = tid >> 5, lane = tid & 31;
  const int hi = lane >> 4, col = lane & 15;
  const int nt0 = wave * 2;

  v8f acc0 = {}, acc1 = {};
  for (int kb = 0; kb < KR; kb += 256) {
    __syncthreads();
    for (int i = tid; i < 16 * 256; i += 256) {
      int r = i >> 8, kkk = i & 255;
      int row = mt * 16 + r;
      _Float16 v = (_Float16)0.f;
      if (row < MR) {
        int k = kb + kkk;
        int c = k / MM, m = k - c * MM;             // channel-major flatten
        int b2 = row / NN, p = row - b2 * NN;
        v = featbuf[(size_t)(b2 * NP + p * MM + m) * CC + c];
      }
      sA2[i] = v;
    }
    __syncthreads();
    __builtin_prefetch(Wr1h + (size_t)(nt0 * 16 + col) * KR + kb + 256, 0, 0);
#pragma unroll
    for (int ks = 0; ks < 8; ++ks) {
      v16h a = load_a_frag(sA2, 0, 256, ks * 32, lane);
      v16h b0 = load_b_frag(Wr1h, (nt0 + 0) * 16, KR, kb + ks * 32, lane);
      acc0 = wmma_f16(a, b0, acc0);
      v16h b1 = load_b_frag(Wr1h, (nt0 + 1) * 16, KR, kb + ks * 32, lane);
      acc1 = wmma_f16(a, b1, acc1);
    }
  }
#pragma unroll
  for (int v = 0; v < 8; ++v) {
    int row = mt * 16 + v + 8 * hi;  // rows >= 200 get zeros (A was zeroed)
    out1h[(size_t)row * ORD + (nt0 + 0) * 16 + col] = (_Float16)fmaxf(acc0[v], 0.f);
    out1h[(size_t)row * ORD + (nt0 + 1) * 16 + col] = (_Float16)fmaxf(acc1[v], 0.f);
  }
}

// ---- 6) reduction MLP layer 2: (200 x 256) @ (256 x 256) -> f32 out -------
__global__ __launch_bounds__(256) void reduce_mlp2_kernel(
    const _Float16* __restrict__ out1h, const _Float16* __restrict__ Wr2h,
    float* __restrict__ out) {
  const int tid = threadIdx.x;
  const int wave = tid >> 5, lane = tid & 31;
  const int t = blockIdx.x * 8 + wave;  // 0..207
  const int mt = t >> 4, nt = t & 15;
  const int hi = lane >> 4, col = lane & 15;
  v8f acc = {};
#pragma unroll
  for (int ks = 0; ks < 8; ++ks) {
    v16h a = load_a_frag(out1h, mt * 16, ORD, ks * 32, lane);
    v16h b = load_b_frag(Wr2h,  nt * 16, ORD, ks * 32, lane);
    acc = wmma_f16(a, b, acc);
  }
#pragma unroll
  for (int v = 0; v < 8; ++v) {
    int row = mt * 16 + v + 8 * hi;
    if (row < MR) out[(size_t)row * ORD + nt * 16 + col] = fmaxf(acc[v], 0.f);
  }
}

// ---- launcher --------------------------------------------------------------
extern "C" void kernel_launch(void* const* d_in, const int* in_sizes, int n_in,
                              void* d_out, int out_size, void* d_ws, size_t ws_size,
                              hipStream_t stream) {
  (void)in_sizes; (void)n_in; (void)out_size;
  const float* proposals = (const float*)d_in[0];
  const float* kxyz      = (const float*)d_in[1];
  const float* kfeat     = (const float*)d_in[2];
  const float* noise     = (const float*)d_in[3];
  const float* W1_0      = (const float*)d_in[4];
  const float* W2_0      = (const float*)d_in[5];
  const float* W1_1      = (const float*)d_in[6];
  const float* W2_1      = (const float*)d_in[7];
  const float* Wr1       = (const float*)d_in[8];
  const float* Wr2       = (const float*)d_in[9];
  float* out = (float*)d_out;

  char* ws = (char*)d_ws;
  size_t o = 0;
  auto alloc = [&](size_t bytes) {
    void* p = ws + o;
    o += (bytes + 255) & ~(size_t)255;
    return p;
  };
  float*    gridpoints = (float*)   alloc((size_t)GT * 3 * 4);
  int*      idx0       = (int*)     alloc((size_t)GT * 16 * 4);
  int*      idx1       = (int*)     alloc((size_t)GT * 16 * 4);
  _Float16* featbuf    = (_Float16*)alloc((size_t)GT * CC * 2);
  _Float16* Wr1h       = (_Float16*)alloc((size_t)ORD * KR * 2);
  _Float16* Wr2h       = (_Float16*)alloc((size_t)ORD * ORD * 2);
  _Float16* out1h      = (_Float16*)alloc((size_t)MRP * ORD * 2);
  if (o > ws_size) return;  // workspace too small: do nothing (deterministic)

  gridpoints_kernel<<<(GT + 255) / 256, 256, 0, stream>>>(proposals, noise, gridpoints);
  cvt_kernel<<<1024, 256, 0, stream>>>(Wr1, Wr1h, ORD * KR);
  cvt_kernel<<<64, 256, 0, stream>>>(Wr2, Wr2h, ORD * ORD);
  dim3 bqGrid((NP + 255) / 256, BB);
  ball_query_kernel<<<bqGrid, 256, 0, stream>>>(gridpoints, kxyz, idx0, idx1);
  group_mlp_kernel<<<GT / 4, 256, 0, stream>>>(gridpoints, kxyz, kfeat, idx0, W1_0, W2_0, featbuf, 0);
  group_mlp_kernel<<<GT / 4, 256, 0, stream>>>(gridpoints, kxyz, kfeat, idx1, W1_1, W2_1, featbuf, 64);
  reduce_mlp1_kernel<<<13, 256, 0, stream>>>(featbuf, Wr1h, out1h);
  reduce_mlp2_kernel<<<26, 256, 0, stream>>>(out1h, Wr2h, out);
}
